// TacotronSTFT_39840116638045
// MI455X (gfx1250) — matrix operand order repaired and checked by reading, not compiled, not run on hardware
//
#include <hip/hip_runtime.h>
#include <hip/hip_bf16.h>

// ---------------- problem constants ----------------
#define FILT     1024
#define HOP      256
#define CUTOFF   513              // FILT/2 + 1
#define NMELS    80
#define BATCH    16
#define NSAMP    524288
#define YPADLEN  (NSAMP + FILT)   // 525312 (reflect pad 512 each side)
#define TFRAMES  2049             // (YPADLEN - FILT)/HOP + 1
#define TPADM    2064             // mag storage stride (multiple of 16)
#define MPAIRS   65               // frames padded to 65*32 = 2080, 2 M-tiles per wave
#define NPAD     1040             // 1026 channels padded to 65*16
#define NTILES   65
#define NGPW     5                // n-tiles per wave  (5*16 = 80 columns)
#define NGROUPS  13               // NTILES / NGPW
// per-batch padded stride for f16 y: must cover frame 2079 -> 2079*256+1024 = 533248
#define YSTRIDE  533504           // rounded up, multiple of 256

typedef __attribute__((ext_vector_type(16))) _Float16 v16h;
typedef __attribute__((ext_vector_type(8)))  _Float16 h8;
typedef __attribute__((ext_vector_type(8)))  float    v8f;

// ---------------- 1) reflect-pad + fp32->fp16 ----------------
__global__ __launch_bounds__(256)
void k_pad(const float* __restrict__ y, _Float16* __restrict__ yp) {
  const int b = blockIdx.y;
  const long i = (long)blockIdx.x * blockDim.x + threadIdx.x;
  if (i >= YSTRIDE) return;
  float v = 0.0f;
  if (i < YPADLEN) {
    long j = i - (FILT / 2);
    if (j < 0)            j = -j;                       // numpy 'reflect' (no edge dup)
    else if (j >= NSAMP)  j = 2L * NSAMP - 2 - j;
    v = y[(long)b * NSAMP + j];
  }
  yp[(long)b * YSTRIDE + i] = (_Float16)v;
}

// ---------------- 2) basis fp32->fp16, real/imag interleaved, padded ----------------
// bb[n][k]: n even -> real channel n/2, n odd -> imag channel n/2; rows 1026..1039 zero.
__global__ __launch_bounds__(256)
void k_basis(const float* __restrict__ fb, _Float16* __restrict__ bb) {
  const long i = (long)blockIdx.x * blockDim.x + threadIdx.x;   // over NPAD*FILT
  if (i >= (long)NPAD * FILT) return;
  const int n = (int)(i >> 10);
  const int k = (int)(i & (FILT - 1));
  float v = 0.0f;
  if (n < 2 * CUTOFF) {
    const int row = (n & 1) * CUTOFF + (n >> 1);        // real rows 0..512, imag rows 513..1025
    v = fb[(long)row * FILT + k];
  }
  bb[i] = (_Float16)v;
}

// ---------------- 3) STFT GEMM (f16 WMMA, f32 acc) + fused magnitude ----------------
// One wave computes a 32(frames) x 80(interleaved channels) tile:
// 2 M-tiles share the 5 B fragments (halves B traffic, 1.4 vmem/wmma).
__global__ __launch_bounds__(128)
void k_stft(const _Float16* __restrict__ yp, const _Float16* __restrict__ bb,
            float* __restrict__ mag) {
  const int lane = threadIdx.x & 31;
  const int wid  = blockIdx.x * 4 + (threadIdx.x >> 5);     // grid sized exactly
  const int ng = wid % NGROUPS;
  const int mp = (wid / NGROUPS) % MPAIRS;                  // pair of M-tiles
  const int b  = wid / (NGROUPS * MPAIRS);

  const int m    = lane & 15;
  const int hsel = lane >> 4;                               // 0: low K-halves, 1: high

  // A: lane holds row m, halves [k0 + 8*hsel .. +7] and [k0 + 8*hsel + 16 .. +7]
  const _Float16* pa0 = yp + (long)b * YSTRIDE + (long)(mp * 32 + m) * HOP + hsel * 8;
  const _Float16* pa1 = pa0 + 16 * HOP;

  // B: lane holds column n, halves [k0 + 8*hsel .. +7] and [k0 + 8*hsel + 16 .. +7]
  const _Float16* pb[NGPW];
#pragma unroll
  for (int g = 0; g < NGPW; ++g) {
    const int n = (ng * NGPW + g) * 16 + m;
    pb[g] = bb + (long)n * FILT + hsel * 8;
  }

  v8f acc[2][NGPW] = {};

  for (int k0 = 0; k0 < FILT; k0 += 32) {
    union { v16h v; h8 h[2]; } a0, a1;
    a0.h[0] = *(const h8*)(pa0 + k0);
    a0.h[1] = *(const h8*)(pa0 + k0 + 16);
    a1.h[0] = *(const h8*)(pa1 + k0);
    a1.h[1] = *(const h8*)(pa1 + k0 + 16);
#pragma unroll
    for (int g = 0; g < NGPW; ++g) {
      union { v16h v; h8 h[2]; } bm;
      bm.h[0] = *(const h8*)(pb[g] + k0);
      bm.h[1] = *(const h8*)(pb[g] + k0 + 16);
      acc[0][g] = __builtin_amdgcn_wmma_f32_16x16x32_f16(
          false, a0.v, false, bm.v, (short)0, acc[0][g], false, false);
      acc[1][g] = __builtin_amdgcn_wmma_f32_16x16x32_f16(
          false, a1.v, false, bm.v, (short)0, acc[1][g], false, false);
    }
  }

  // C layout: lane holds column n = n0 + (lane&15); VGPR j holds row M = j + 8*hsel.
  // Adjacent lanes hold (real, imag) of the same channel -> shfl_xor(1) + sqrt.
#pragma unroll
  for (int mi = 0; mi < 2; ++mi) {
    const int tBase = mp * 32 + mi * 16;
#pragma unroll
    for (int g = 0; g < NGPW; ++g) {
      const int  n      = (ng * NGPW + g) * 16 + m;
      const int  c      = n >> 1;
      const bool isReal = (n & 1) == 0;
#pragma unroll
      for (int j = 0; j < 8; ++j) {
        const float self  = acc[mi][g][j];
        const float other = __shfl_xor(self, 1, 32);        // partner (imag) lane
        const int   t     = tBase + j + hsel * 8;
        if (isReal && c < CUTOFF && t < TFRAMES) {
          mag[((long)b * CUTOFF + c) * TPADM + t] = sqrtf(self * self + other * other);
        }
      }
    }
  }
}

// ---------------- 4) mel projection + log ----------------
// grid: (t-chunks=9, mel-groups of 8 = 10, batch=16); coalesced mag reads over t.
__global__ __launch_bounds__(256)
void k_mel(const float* __restrict__ melw, const float* __restrict__ mag,
           float* __restrict__ out) {
  const int tc = blockIdx.x, mg = blockIdx.y, b = blockIdx.z;
  __shared__ float w[8][CUTOFF];
  for (int i = threadIdx.x; i < 8 * CUTOFF; i += 256) {
    const int j = i / CUTOFF, c = i - j * CUTOFF;
    w[j][c] = melw[(mg * 8 + j) * CUTOFF + c];
  }
  __syncthreads();
  const int t = tc * 256 + threadIdx.x;
  if (t >= TFRAMES) return;
  float acc[8] = {};
  const float* mp = mag + (long)b * CUTOFF * TPADM + t;
  for (int c = 0; c < CUTOFF; ++c) {
    const float v = mp[(long)c * TPADM];
#pragma unroll
    for (int j = 0; j < 8; ++j) acc[j] = fmaf(w[j][c], v, acc[j]);
  }
#pragma unroll
  for (int j = 0; j < 8; ++j) {
    out[((long)b * NMELS + (mg * 8 + j)) * TFRAMES + t] = logf(fmaxf(acc[j], 1e-5f));
  }
}

// ---------------- launch ----------------
extern "C" void kernel_launch(void* const* d_in, const int* in_sizes, int n_in,
                              void* d_out, int out_size, void* d_ws, size_t ws_size,
                              hipStream_t stream) {
  const float* y  = (const float*)d_in[0];   // (16, 524288)
  const float* fb = (const float*)d_in[1];   // (1026, 1024)
  const float* mw = (const float*)d_in[2];   // (80, 513)
  float* out = (float*)d_out;                // (16, 80, 2049)

  char* ws = (char*)d_ws;
  _Float16* yp = (_Float16*)ws;                                   // 16*533504*2  ≈ 17.1 MB
  size_t off1 = (((size_t)BATCH * YSTRIDE * 2) + 255) & ~(size_t)255;
  _Float16* bb = (_Float16*)(ws + off1);                          // 1040*1024*2  ≈ 2.1 MB
  size_t off2 = ((off1 + (size_t)NPAD * FILT * 2) + 255) & ~(size_t)255;
  float* mag = (float*)(ws + off2);                               // 16*513*2064*4 ≈ 67.8 MB
  (void)in_sizes; (void)n_in; (void)out_size; (void)ws_size;

  k_pad  <<<dim3((YSTRIDE + 255) / 256, BATCH), 256, 0, stream>>>(y, yp);
  k_basis<<<((NPAD * FILT) + 255) / 256,        256, 0, stream>>>(fb, bb);

  const int totalWaves = BATCH * MPAIRS * NGROUPS;                // 13520, 4 waves per block
  k_stft <<<totalWaves / 4, 128, 0, stream>>>(yp, bb, mag);

  k_mel  <<<dim3((TFRAMES + 255) / 256, NMELS / 8, BATCH), 256, 0, stream>>>(mw, mag, out);
}
